// VectorQuantizer_17927193494119
// MI455X (gfx1250) — compile-verified, matrix operand order checked by reading, not run on hardware
//
#include <hip/hip_runtime.h>
#include <hip/hip_bf16.h>
#include <stdint.h>

typedef __attribute__((ext_vector_type(16))) __bf16 v16bf;
typedef __attribute__((ext_vector_type(8)))  float  v8f;
typedef __attribute__((ext_vector_type(4)))  float  v4f;

#define VQ_D   64
#define VQ_K   1024
#define NTOK   (16 * 8192)        // B*T = 131072
#define NTILE  (NTOK / 16)        // 8192 token tiles of 16

// d_ws layout:
//   [0,      131072)  wh fragments (65536 bf16)  -- B-frag register order
//   [131072, 262144)  wl fragments (65536 bf16)
//   [262144, 266240)  he2 = 0.5*||e_n||^2 (1024 f32)

// Split weight into bf16 hi/lo, stored directly in the WMMA B-fragment layout:
// flat index t -> j = t&15 (element in v16bf), lane = (t>>4)&31, khalf = (t>>9)&1,
// chunk c = t>>10.  n = c*16 + (lane&15), k = 32*khalf + (j<8 ? j+8g : j+8+8g), g=lane>>4.
__global__ void vq_prep_frags(const float* __restrict__ w,
                              __bf16* __restrict__ whf,
                              __bf16* __restrict__ wlf) {
    int t    = blockIdx.x * blockDim.x + threadIdx.x;   // 0..65535
    int j    = t & 15;
    int lane = (t >> 4) & 31;
    int h2   = (t >> 9) & 1;
    int c    = t >> 10;
    int n    = c * 16 + (lane & 15);
    int g    = lane >> 4;
    int k    = 32 * h2 + ((j < 8) ? (j + 8 * g) : (j + 8 + 8 * g));
    float v  = w[n * VQ_D + k];
    __bf16 hi = (__bf16)v;
    __bf16 lo = (__bf16)(v - (float)hi);
    whf[t] = hi;
    wlf[t] = lo;
}

__global__ void vq_prep_he2(const float* __restrict__ w, float* __restrict__ he2) {
    int n = blockIdx.x * blockDim.x + threadIdx.x;      // 0..1023
    float s = 0.f;
    #pragma unroll
    for (int k = 0; k < VQ_D; ++k) { float v = w[n * VQ_D + k]; s += v * v; }
    he2[n] = 0.5f * s;
}

__global__ __launch_bounds__(256)
void vq_main(const float* __restrict__ x,
             const float* __restrict__ weight,
             const v16bf* __restrict__ whf,
             const v16bf* __restrict__ wlf,
             const float* __restrict__ he2,
             float* __restrict__ outQ,
             float* __restrict__ outI) {
    const int lane  = threadIdx.x & 31;
    const int wave  = threadIdx.x >> 5;
    const int tile0 = (blockIdx.x * 8 + wave) * 2;      // this wave: tiles tile0, tile0+1
    const int m     = lane & 15;
    const int g     = lane >> 4;

    // ---- load two 16x64 x-tiles, split to bf16 hi/lo A-fragments ----
    // A 16x32 bf16 layout: lane holds row m=lane&15; element j -> k = (j<8 ? j+8g : j+8+8g)
    v16bf ah[2][2], al[2][2];                           // [tile][khalf]
    #pragma unroll
    for (int tt = 0; tt < 2; ++tt) {
        const float* xrow = x + (size_t)((tile0 + tt) * 16 + m) * VQ_D;
        float xa[32];
        #pragma unroll
        for (int h2 = 0; h2 < 2; ++h2) {
            const v4f* p0 = (const v4f*)(xrow + 32 * h2 + 8 * g);
            const v4f* p1 = (const v4f*)(xrow + 32 * h2 + 16 + 8 * g);
            v4f q0 = p0[0], q1 = p0[1], q2 = p1[0], q3 = p1[1];
            #pragma unroll
            for (int e = 0; e < 4; ++e) {
                xa[h2 * 16 + 0  + e] = q0[e];
                xa[h2 * 16 + 4  + e] = q1[e];
                xa[h2 * 16 + 8  + e] = q2[e];
                xa[h2 * 16 + 12 + e] = q3[e];
            }
        }
        #pragma unroll
        for (int h2 = 0; h2 < 2; ++h2) {
            #pragma unroll
            for (int j = 0; j < 16; ++j) {
                float v   = xa[h2 * 16 + j];
                __bf16 hi = (__bf16)v;
                ah[tt][h2][j] = hi;
                al[tt][h2][j] = (__bf16)(v - (float)hi);
            }
        }
    }

    float best[2][8];
    int   bidx[2][8];
    #pragma unroll
    for (int tt = 0; tt < 2; ++tt)
        #pragma unroll
        for (int r = 0; r < 8; ++r) { best[tt][r] = 3.4e38f; bidx[tt][r] = 0; }

    // ---- 64 chunks of 16 codes: one B-fragment load feeds two tiles' WMMA chains ----
    #pragma unroll 2
    for (int c = 0; c < 64; ++c) {
        int fi0 = (c * 2) * 32 + lane;
        int fi1 = fi0 + 32;
        v16bf bh0 = whf[fi0], bh1 = whf[fi1];
        v16bf bl0 = wlf[fi0], bl1 = wlf[fi1];

        int   n = c * 16 + m;
        float h = he2[n];

        v8f acc[2];
        #pragma unroll
        for (int tt = 0; tt < 2; ++tt) {
            v8f a = {0.f, 0.f, 0.f, 0.f, 0.f, 0.f, 0.f, 0.f};
            a = __builtin_amdgcn_wmma_f32_16x16x32_bf16(false, ah[tt][0], false, bh0, (short)0, a, false, false);
            a = __builtin_amdgcn_wmma_f32_16x16x32_bf16(false, ah[tt][1], false, bh1, (short)0, a, false, false);
            a = __builtin_amdgcn_wmma_f32_16x16x32_bf16(false, ah[tt][0], false, bl0, (short)0, a, false, false);
            a = __builtin_amdgcn_wmma_f32_16x16x32_bf16(false, al[tt][0], false, bh0, (short)0, a, false, false);
            a = __builtin_amdgcn_wmma_f32_16x16x32_bf16(false, ah[tt][1], false, bl1, (short)0, a, false, false);
            a = __builtin_amdgcn_wmma_f32_16x16x32_bf16(false, al[tt][1], false, bh1, (short)0, a, false, false);
            acc[tt] = a;
        }

        #pragma unroll
        for (int tt = 0; tt < 2; ++tt) {
            #pragma unroll
            for (int r = 0; r < 8; ++r) {
                float d = h - acc[tt][r];             // argmin(0.5||e||^2 - x.e)
                if (d < best[tt][r]) { best[tt][r] = d; bidx[tt][r] = n; }
            }
        }
    }

    // ---- reduce argmin across the 16 lanes sharing each token row ----
    #pragma unroll
    for (int mask = 1; mask <= 8; mask <<= 1) {
        #pragma unroll
        for (int tt = 0; tt < 2; ++tt) {
            #pragma unroll
            for (int r = 0; r < 8; ++r) {
                float ob = __shfl_xor(best[tt][r], mask, 32);
                int   oi = __shfl_xor(bidx[tt][r], mask, 32);
                if (ob < best[tt][r] || (ob == best[tt][r] && oi < bidx[tt][r])) {
                    best[tt][r] = ob; bidx[tt][r] = oi;
                }
            }
        }
    }

    // ---- epilogue: token mt lives at slot r=mt&7 in lane-half g=mt>>3 ----
    const float2* w2 = (const float2*)weight;
    float2*       q2 = (float2*)outQ;
    #pragma unroll
    for (int tt = 0; tt < 2; ++tt) {
        const int tile = tile0 + tt;
        #pragma unroll
        for (int mt = 0; mt < 16; ++mt) {
            int idx = __shfl(bidx[tt][mt & 7], (mt >> 3) * 16, 32);
            if (lane == mt) outI[tile * 16 + mt] = (float)idx;
            q2[(size_t)(tile * 16 + mt) * 32 + lane] = w2[(size_t)idx * 32 + lane];
        }
    }
}

extern "C" void kernel_launch(void* const* d_in, const int* in_sizes, int n_in,
                              void* d_out, int out_size, void* d_ws, size_t ws_size,
                              hipStream_t stream) {
    const float* x = (const float*)d_in[0];   // [B,T,D,1] f32
    const float* w = (const float*)d_in[1];   // [K,D] f32

    char*   ws  = (char*)d_ws;
    __bf16* whf = (__bf16*)ws;
    __bf16* wlf = (__bf16*)(ws + 131072);
    float*  he2 = (float*)(ws + 262144);

    float* outQ = (float*)d_out;              // [B,T,D] f32
    float* outI = outQ + (size_t)NTOK * VQ_D; // [B,T] indices (as float)

    vq_prep_frags<<<65536 / 256, 256, 0, stream>>>(w, whf, wlf);
    vq_prep_he2<<<VQ_K / 256, 256, 0, stream>>>(w, he2);
    // 2 tiles per wave, 8 waves per block -> NTILE/16 blocks
    vq_main<<<NTILE / 16, 256, 0, stream>>>(x, w, (const v16bf*)whf, (const v16bf*)wlf,
                                            he2, outQ, outI);
}
